// SparseConvNeXtV2_21260088115535
// MI455X (gfx1250) — compile-verified, matrix operand order checked by reading, not run on hardware
//
#include <hip/hip_runtime.h>
#include <hip/hip_bf16.h>
#include <math.h>

// ---------------------------------------------------------------------------
// Types for CDNA5 WMMA (gfx1250): bf16 16x16x32, f32 accumulate
// ---------------------------------------------------------------------------
typedef __attribute__((ext_vector_type(16))) __bf16 v16bf;
typedef __attribute__((ext_vector_type(8)))  float  v8f;

union Frag16 { v16bf v; unsigned short s[16]; uint4 q[2]; };

__device__ __forceinline__ unsigned short f32_to_bf16(float f) {
    unsigned int u = __float_as_uint(f);
    unsigned int r = 0x7FFFu + ((u >> 16) & 1u);   // round to nearest even
    return (unsigned short)((u + r) >> 16);
}

// keep mask: mask is (16, 256) int32; keep_i[b,t] = 1 - mask[b][(t<<mshift)>>3]
// L is always a power of two -> pass log2(L), no integer division.
__device__ __forceinline__ float keepf(const int* __restrict__ mask, int row,
                                       int Lshift, int mshift) {
    int b = row >> Lshift;
    int t = row & ((1 << Lshift) - 1);
    int mi = (t << mshift) >> 3;
    return 1.0f - (float)mask[(b << 8) + mi];
}

// ---------------------------------------------------------------------------
// Stem: h[b,t,c] = LN( sum_k x[b,4t+k]*w[k,c] + bias )[c] * keep   (C = 96)
// one wave per (b,t) row; L = 2048 (Lshift = 11)
// ---------------------------------------------------------------------------
__global__ __launch_bounds__(256)
void stem_kernel(const float* __restrict__ x, float* __restrict__ h,
                 const float* __restrict__ w, const float* __restrict__ bias,
                 const float* __restrict__ lnw, const float* __restrict__ lnb,
                 const int* __restrict__ mask) {
    const int Lshift = 11;
    int row  = blockIdx.x * 8 + (threadIdx.x >> 5);
    int lane = threadIdx.x & 31;
    int b = row >> Lshift, t = row & 2047;
    const float* xb = x + (size_t)b * 8192 + (size_t)t * 4;
    float x0 = xb[0], x1 = xb[1], x2 = xb[2], x3 = xb[3];

    float vals[3], s = 0.f, s2 = 0.f;
#pragma unroll
    for (int i = 0; i < 3; ++i) {
        int c = lane + 32 * i;
        float v = x0 * w[c] + x1 * w[96 + c] + x2 * w[192 + c] + x3 * w[288 + c] + bias[c];
        vals[i] = v; s += v; s2 += v * v;
    }
#pragma unroll
    for (int off = 16; off; off >>= 1) {
        s  += __shfl_xor(s,  off, 32);
        s2 += __shfl_xor(s2, off, 32);
    }
    float mu  = s * (1.0f / 96.0f);
    float var = s2 * (1.0f / 96.0f) - mu * mu;
    float inv = rsqrtf(var + 1e-6f);
    float kp  = keepf(mask, row, Lshift, 0);
    float* hr = h + (size_t)row * 96;
#pragma unroll
    for (int i = 0; i < 3; ++i) {
        int c = lane + 32 * i;
        hr[c] = ((vals[i] - mu) * inv * lnw[c] + lnb[c]) * kp;
    }
}

// ---------------------------------------------------------------------------
// LayerNorm(+mask): out = LN(in)*keep   (wave per row, lanes stride channels)
// ---------------------------------------------------------------------------
__global__ __launch_bounds__(256)
void ln_mask_kernel(const float* __restrict__ in, float* __restrict__ out,
                    const float* __restrict__ w, const float* __restrict__ b,
                    const int* __restrict__ mask, int mshift, int Lshift, int C) {
    int row  = blockIdx.x * 8 + (threadIdx.x >> 5);
    int lane = threadIdx.x & 31;
    const float* ir = in + (size_t)row * C;
    float s = 0.f, s2 = 0.f;
    for (int c = lane; c < C; c += 32) { float v = ir[c]; s += v; s2 += v * v; }
#pragma unroll
    for (int off = 16; off; off >>= 1) {
        s  += __shfl_xor(s,  off, 32);
        s2 += __shfl_xor(s2, off, 32);
    }
    float mu  = s / (float)C;
    float var = s2 / (float)C - mu * mu;
    float inv = rsqrtf(var + 1e-6f);
    float kp  = keepf(mask, row, Lshift, mshift);
    float* orow = out + (size_t)row * C;
    for (int c = lane; c < C; c += 32)
        orow[c] = ((ir[c] - mu) * inv * w[c] + b[c]) * kp;
}

// ---------------------------------------------------------------------------
// Depthwise conv (K=7, zero pad 3) + LayerNorm + mask, fused. Wave per row.
// ---------------------------------------------------------------------------
__global__ __launch_bounds__(256)
void dwln_kernel(const float* __restrict__ h, float* __restrict__ out,
                 const float* __restrict__ dww, const float* __restrict__ dwb,
                 const float* __restrict__ lnw, const float* __restrict__ lnb,
                 const int* __restrict__ mask, int mshift, int Lshift, int C) {
    int row  = blockIdx.x * 8 + (threadIdx.x >> 5);
    int lane = threadIdx.x & 31;
    int L = 1 << Lshift;
    int b = row >> Lshift, t = row & (L - 1);
    const float* hb = h + ((size_t)b << Lshift) * C;

    float s = 0.f, s2 = 0.f;
    for (int c = lane; c < C; c += 32) {
        float v = dwb[c];
#pragma unroll
        for (int k = 0; k < 7; ++k) {
            int tt = t + k - 3;
            if (tt >= 0 && tt < L) v += hb[(size_t)tt * C + c] * dww[k * C + c];
        }
        s += v; s2 += v * v;
    }
#pragma unroll
    for (int off = 16; off; off >>= 1) {
        s  += __shfl_xor(s,  off, 32);
        s2 += __shfl_xor(s2, off, 32);
    }
    float mu  = s / (float)C;
    float var = s2 / (float)C - mu * mu;
    float inv = rsqrtf(var + 1e-6f);
    float kp  = keepf(mask, row, Lshift, mshift);
    float* orow = out + (size_t)row * C;
    for (int c = lane; c < C; c += 32) {
        float v = dwb[c];
#pragma unroll
        for (int k = 0; k < 7; ++k) {
            int tt = t + k - 3;
            if (tt >= 0 && tt < L) v += hb[(size_t)tt * C + c] * dww[k * C + c];
        }
        orow[c] = ((v - mu) * inv * lnw[c] + lnb[c]) * kp;
    }
}

// ---------------------------------------------------------------------------
// Pack bf16 weights into WMMA B-fragment-major layout:
//   Wp[((kt*ntilesN + nt)*32 + lane)*16 + e]
// so each lane fetches its 16 fragment values with two global_load_b128.
// Fragment mapping (ISA 7.12.2, 16-bit B 32x16): lane&15 = n within tile,
// lane>>4 = khalf; e<8 -> k = khalf*8+e ; e>=8 -> k = 16+khalf*8+(e-8).
// ---------------------------------------------------------------------------
__global__ __launch_bounds__(256)
void pack_w_kernel(const float* __restrict__ W, unsigned short* __restrict__ Wp,
                   int N, int ntilesN, int total) {
    int idx = blockIdx.x * 256 + threadIdx.x;
    if (idx >= total) return;
    int e    = idx & 15;
    int l    = (idx >> 4) & 31;
    int tile = idx >> 9;
    int nt   = tile % ntilesN;
    int kt   = tile / ntilesN;
    int n     = nt * 16 + (l & 15);
    int khalf = l >> 4;
    int k     = kt * 32 + ((e < 8) ? (khalf * 8 + e) : (16 + khalf * 8 + (e - 8)));
    Wp[idx] = f32_to_bf16(W[(size_t)k * N + n]);
}

// ---------------------------------------------------------------------------
// WMMA GEMM: D[M,N] = A[M,K] @ W[K,N] with fused epilogues.
//   mode 0 (downsample): out = (acc + bias) * keep
//   mode 1 (mlp1)      : y = gelu(acc + bias) * keep ; out=y ; sumsq[n]+=sum y^2
//   mode 2 (mlp2)      : A' = (gamma*nx*A + beta + A)*keep (staged) ;
//                        out = hres + (acc + bias) * keep
// 8 waves/block; each wave owns TWO adjacent 16x16 N-tiles sharing one A
// fragment (2 back-to-back v_wmma per K-step). A tile staged to LDS already
// converted to bf16 in fragment layout -> 2x ds_load_b128 per lane. W is
// pre-packed fragment-major -> 2x global_load_b128 per tile per K-step.
// ---------------------------------------------------------------------------
__global__ __launch_bounds__(256)
void gemm_kernel(const float* __restrict__ A, const unsigned short* __restrict__ W,
                 const float* __restrict__ bias, float* __restrict__ out,
                 const float* __restrict__ hres, float* __restrict__ sumsq,
                 const float* __restrict__ gamma, const float* __restrict__ beta,
                 const float* __restrict__ nx, const int* __restrict__ mask,
                 int mshift, int Lshift, int K, int N, int mode) {
    __shared__ uint4 AfragLds[64];                 // 32 lanes x 32B (bf16 A fragment)
    unsigned short* ABs = reinterpret_cast<unsigned short*>(AfragLds);

    int tid     = threadIdx.x;
    int lane    = tid & 31;
    int wid     = tid >> 5;
    int row0    = blockIdx.x * 16;
    int ntilesN = N >> 4;
    int nt0     = blockIdx.y * 16 + wid * 2;       // first of two N-tiles
    int nt1     = nt0 + 1;
    bool ok0    = (nt0 < ntilesN);
    bool ok1    = (nt1 < ntilesN);

    int khalf = lane >> 4;
    v8f acc0 = {0.f, 0.f, 0.f, 0.f, 0.f, 0.f, 0.f, 0.f};
    v8f acc1 = acc0;

    const uint4* Wq = reinterpret_cast<const uint4*>(W);

    for (int kt = 0; kt * 32 < K; ++kt) {
        int k0 = kt * 32;
        // ---- stage A tile straight into bf16 fragment layout ----
        for (int idx = tid; idx < 512; idx += 256) {
            int e  = idx & 15;
            int l  = idx >> 4;
            int m  = l & 15;
            int kh = l >> 4;
            int k  = (e < 8) ? (kh * 8 + e) : (16 + kh * 8 + (e - 8));
            int row = row0 + m;
            float v = A[(size_t)row * K + (k0 + k)];
            if (mode == 2) {
                int kk = k0 + k;
                float kp = keepf(mask, row, Lshift, mshift);
                v = (gamma[kk] * (v * nx[kk]) + beta[kk] + v) * kp;
            }
            ABs[idx] = f32_to_bf16(v);
        }
        if (k0 + 32 < K)
            __builtin_prefetch(&A[(size_t)(row0 + (tid & 15)) * K + k0 + 32], 0, 1);
        __syncthreads();

        Frag16 af;
        af.q[0] = AfragLds[lane * 2];
        af.q[1] = AfragLds[lane * 2 + 1];

        if (ok0) {
            Frag16 bf;
            size_t wb = (((size_t)kt * ntilesN + nt0) * 32 + lane) * 2;
            bf.q[0] = Wq[wb];
            bf.q[1] = Wq[wb + 1];
            acc0 = __builtin_amdgcn_wmma_f32_16x16x32_bf16(
                false, af.v, false, bf.v, (short)0, acc0, false, false);
        }
        if (ok1) {
            Frag16 bf;
            size_t wb = (((size_t)kt * ntilesN + nt1) * 32 + lane) * 2;
            bf.q[0] = Wq[wb];
            bf.q[1] = Wq[wb + 1];
            acc1 = __builtin_amdgcn_wmma_f32_16x16x32_bf16(
                false, af.v, false, bf.v, (short)0, acc1, false, false);
        }
        __syncthreads();
    }

    // ---- epilogue ----
#pragma unroll
    for (int tile = 0; tile < 2; ++tile) {
        int nt = tile ? nt1 : nt0;
        if (nt >= ntilesN) continue;
        v8f acc = tile ? acc1 : acc0;
        int n = nt * 16 + (lane & 15);
        float bv = bias[n];
        float partial = 0.f;
#pragma unroll
        for (int r = 0; r < 8; ++r) {
            int row  = row0 + r + 8 * khalf;   // D layout: VGPR r -> M=r (+8 hi lanes)
            float kp = keepf(mask, row, Lshift, mshift);
            float v  = acc[r] + bv;
            size_t o = (size_t)row * N + n;
            if (mode == 1) {
                v = 0.5f * v * (1.0f + erff(v * 0.70710678118654752f)) * kp;  // exact GELU
                partial += v * v;
                out[o] = v;
            } else if (mode == 0) {
                out[o] = v * kp;
            } else {
                out[o] = hres[o] + v * kp;
            }
        }
        if (mode == 1) atomicAdd(&sumsq[n], partial);
    }
}

// ---------------------------------------------------------------------------
// GRN finalize: Nx[c] = sqrt(sumsq[c]) / (mean_c sqrt(sumsq) + 1e-6)
// ---------------------------------------------------------------------------
__global__ __launch_bounds__(256)
void grn_kernel(const float* __restrict__ sumsq, float* __restrict__ nxout, int C4) {
    __shared__ float red[256];
    float loc = 0.f;
    for (int c = threadIdx.x; c < C4; c += 256) loc += sqrtf(sumsq[c]);
    red[threadIdx.x] = loc;
    __syncthreads();
    for (int s = 128; s; s >>= 1) {
        if (threadIdx.x < s) red[threadIdx.x] += red[threadIdx.x + s];
        __syncthreads();
    }
    float denom = red[0] / (float)C4 + 1e-6f;
    for (int c = threadIdx.x; c < C4; c += 256)
        nxout[c] = sqrtf(sumsq[c]) / denom;
}

__global__ __launch_bounds__(256)
void zero_kernel(float* __restrict__ p, int nelem) {
    int i = blockIdx.x * 256 + threadIdx.x;
    if (i < nelem) p[i] = 0.f;
}

// final (B, L, C) -> (B, C, L);  L = 256, C = 768
__global__ __launch_bounds__(256)
void transpose_kernel(const float* __restrict__ h, float* __restrict__ out) {
    int i = blockIdx.x * 256 + threadIdx.x;          // index into out (B, C, L)
    int t = i & 255;
    int c = (i >> 8) % 768;
    int b = i / (256 * 768);
    out[i] = h[(((size_t)b << 8) + t) * 768 + c];
}

// ---------------------------------------------------------------------------
// Orchestration
// d_in order (pytree leaves, insertion order):
//  0:x  1:mask
//  2..5   stem   {w, b, ln_w, ln_b}
//  6..17  down[i]{ln_w, ln_b, w, b}          i = 0..2
//  18..   block  {dw_w, dw_b, ln_w, ln_b, w1, b1, gamma, beta, w2, b2} x18
// ---------------------------------------------------------------------------
extern "C" void kernel_launch(void* const* d_in, const int* in_sizes, int n_in,
                              void* d_out, int out_size, void* d_ws, size_t ws_size,
                              hipStream_t stream) {
    (void)in_sizes; (void)n_in; (void)out_size; (void)ws_size;
    const float* x    = (const float*)d_in[0];
    const int*   mask = (const int*)d_in[1];
    auto P = [&](int i) { return (const float*)d_in[i]; };

    const int B = 16;
    const int Lsh[4]    = {11, 10, 9, 8};            // log2 of L per stage
    const int Cs[4]     = {96, 192, 384, 768};
    const int depths[4] = {3, 3, 9, 3};
    const int cum[4]    = {0, 3, 6, 15};
    const size_t ACT = (size_t)B * 2048 * 96;        // 3,145,728 (constant per stage)

    float* hbuf  = (float*)d_ws;                     // (B, L, C)
    float* tbuf  = hbuf + ACT;                       // LN / dwconv output
    float* ybuf  = tbuf + ACT;                       // (B, L, 4C) expansion
    float* sumsq = ybuf + 4 * ACT;                   // 3072
    float* nxbuf = sumsq + 3072;                     // 3072
    unsigned short* wreg = (unsigned short*)(nxbuf + 3072);  // packed bf16 weights

    // Stem
    stem_kernel<<<(B * 2048) / 8, 256, 0, stream>>>(x, hbuf, P(2), P(3), P(4), P(5), mask);

    for (int i = 0; i < 4; ++i) {
        int Ls = Lsh[i], C = Cs[i], M = B << Ls, C4 = 4 * C;

        if (i > 0) {
            int Lsp = Lsh[i - 1], Cp = Cs[i - 1], Mp = B << Lsp;
            int base = 6 + 4 * (i - 1);              // {ln_w, ln_b, w, b}
            ln_mask_kernel<<<Mp / 8, 256, 0, stream>>>(hbuf, tbuf, P(base), P(base + 1),
                                                       mask, i - 1, Lsp, Cp);
            int K = 2 * Cp, N = C;                   // stride-2 k=2 conv == GEMM on paired rows
            pack_w_kernel<<<(K * N + 255) / 256, 256, 0, stream>>>(P(base + 2), wreg,
                                                                   N, N >> 4, K * N);
            dim3 g(M / 16, (N + 255) / 256);
            gemm_kernel<<<g, 256, 0, stream>>>(tbuf, wreg, P(base + 3), hbuf,
                                               nullptr, nullptr, nullptr, nullptr, nullptr,
                                               mask, i, Ls, K, N, 0);
        }

        for (int j = 0; j < depths[i]; ++j) {
            int bp = 18 + 10 * (cum[i] + j);
            // dwconv7 + LN + mask
            dwln_kernel<<<M / 8, 256, 0, stream>>>(hbuf, tbuf, P(bp), P(bp + 1),
                                                   P(bp + 2), P(bp + 3), mask, i, Ls, C);
            // GEMM1: C -> 4C, GELU, mask, per-channel sum of squares
            zero_kernel<<<(C4 + 255) / 256, 256, 0, stream>>>(sumsq, C4);
            pack_w_kernel<<<(C * C4 + 255) / 256, 256, 0, stream>>>(P(bp + 4), wreg,
                                                                    C4, C4 >> 4, C * C4);
            dim3 g1(M / 16, (C4 + 255) / 256);
            gemm_kernel<<<g1, 256, 0, stream>>>(tbuf, wreg, P(bp + 5), ybuf,
                                                nullptr, sumsq, nullptr, nullptr, nullptr,
                                                mask, i, Ls, C, C4, 1);
            // GRN global normalizer
            grn_kernel<<<1, 256, 0, stream>>>(sumsq, nxbuf, C4);
            // GEMM2: 4C -> C with GRN transform folded into A staging; residual add
            pack_w_kernel<<<(C4 * C + 255) / 256, 256, 0, stream>>>(P(bp + 8), wreg,
                                                                    C, C >> 4, C4 * C);
            dim3 g2(M / 16, (C + 255) / 256);
            gemm_kernel<<<g2, 256, 0, stream>>>(ybuf, wreg, P(bp + 9), hbuf,
                                                hbuf, nullptr, P(bp + 6), P(bp + 7), nxbuf,
                                                mask, i, Ls, C4, C, 2);
        }
    }

    // (B, 256, 768) -> (B, 768, 256)
    int n = B * 768 * 256;
    transpose_kernel<<<(n + 255) / 256, 256, 0, stream>>>(hbuf, (float*)d_out);
}
